// MultiheadAttention_2044404433352
// MI455X (gfx1250) — matrix operand order repaired and checked by reading, not compiled, not run on hardware
//
#include <hip/hip_runtime.h>
#include <cstdint>
#include <cstddef>

// ---------------------------------------------------------------------------
// MultiheadAttention layer for MI455X (gfx1250), all matmuls on
// v_wmma_f32_16x16x32_bf16.
//   1) cvt q,k,v f32->bf16
//   2) pack Wq/Wk/Wv/Wo into WMMA B-fragment lane order (bf16)
//   3) GEMM (4 row-tiles/wave, B-fragments resident): qp (f32+bf16),
//      kp (bf16), vT = (v@Wv+bv)^T per batch (bf16, [B][D][NK])
//   4) attention: 4 waves cooperate on one (b,h,16-query tile):
//      QK^T WMMA -> masked logits in 128KB LDS slab (padded rows) ->
//      cross-wave 2-pass softmax -> PV WMMA partials -> LDS reduce -> o
//   5) x1 = LN(qp + o);  y = relu(x1@Wo+bo) via WMMA;  out = LN(x1 + y)
// ---------------------------------------------------------------------------

#define B_  4
#define NQ_ 2048
#define NK_ 2048
#define D_  128
#define H_  8
#define DH_ 16

typedef __attribute__((ext_vector_type(16))) __bf16 v16bf;
typedef __attribute__((ext_vector_type(8)))  float  v8f;

union Half8 { uint4 u; __bf16 h[8]; };

__device__ __forceinline__ uint16_t f2bf_bits(float f) {
  union { float f; uint32_t u; } c; c.f = f;
  uint32_t u = c.u;
  u = (u + 0x7FFFu + ((u >> 16) & 1u)) >> 16;   // round-to-nearest-even
  return (uint16_t)u;
}
__device__ __forceinline__ __bf16 f2bf(float f) {
  union { uint16_t s; __bf16 h; } o; o.s = f2bf_bits(f);
  return o.h;
}

// ---------------------------------------------------------------- cvt f32->bf16
__global__ void cvt_f32_bf16(const float* __restrict__ src,
                             uint16_t* __restrict__ dst, int n) {
  int i = blockIdx.x * blockDim.x + threadIdx.x;
  if (i < n) dst[i] = f2bf_bits(src[i]);
}

// ------------------------------------------------- pack W (128x128) B-fragments
// Wpk[((kt*8 + nt)*32 + lane)*16 + e] =
//   bf16(W[kt*32 + (lane>>4)*16 + e][nt*16 + (lane&15)])
__global__ void pack_w(const float* __restrict__ W, uint16_t* __restrict__ Wpk) {
  int idx = blockIdx.x * blockDim.x + threadIdx.x;          // 16384 total
  int e    = idx & 15;
  int lane = (idx >> 4) & 31;
  int nt   = (idx >> 9) & 7;
  int kt   = idx >> 12;
  int row = kt * 32 + (lane >> 4) * 16 + e;
  int col = nt * 16 + (lane & 15);
  Wpk[idx] = f2bf_bits(W[row * D_ + col]);
}

// ----------------------------------------------------------------- bf16 GEMM
// 8 waves/block; wave = column tile (all 8 cover D=128); each wave sweeps
// 4 row tiles (64 rows) with all four K=32 B-fragments kept in registers.
__global__ __launch_bounds__(256)
void gemm_bf16(const uint16_t* __restrict__ Au, const uint16_t* __restrict__ Wpku,
               const float* __restrict__ bias,
               float* __restrict__ outF, uint16_t* __restrict__ outB,
               uint16_t* __restrict__ outT, int tokensPerBatch, int doRelu)
{
  const __bf16* A   = (const __bf16*)Au;
  const __bf16* Wpk = (const __bf16*)Wpku;
  const int lane = threadIdx.x & 31;
  const int wave = threadIdx.x >> 5;            // column tile 0..7
  const int rt0  = blockIdx.x * 4;              // first of 4 row tiles
  const int kb   = (lane >> 4) * 16;

  // Resident B fragments for the whole K=128 strip of this column tile.
  v16bf bf[4];
#pragma unroll
  for (int kt = 0; kt < 4; ++kt) {
    const __bf16* bp = Wpk + (((size_t)(kt * 8 + wave) * 32 + lane) * 16);
    Half8 b0, b1;
    b0.u = *(const uint4*)(bp);
    b1.u = *(const uint4*)(bp + 8);
#pragma unroll
    for (int i = 0; i < 8; ++i) { bf[kt][i] = b0.h[i]; bf[kt][8 + i] = b1.h[i]; }
  }

  v8f acc[4];
#pragma unroll
  for (int r4 = 0; r4 < 4; ++r4) acc[r4] = (v8f){};

#pragma unroll
  for (int r4 = 0; r4 < 4; ++r4) {
    const int mrow = (rt0 + r4) * 16 + (lane & 15);
#pragma unroll
    for (int kt = 0; kt < 4; ++kt) {
      Half8 a0, a1;
      const __bf16* arow = A + (size_t)mrow * D_ + kt * 32 + kb;
      a0.u = *(const uint4*)(arow);
      a1.u = *(const uint4*)(arow + 8);
      v16bf af;
#pragma unroll
      for (int i = 0; i < 8; ++i) { af[i] = a0.h[i]; af[8 + i] = a1.h[i]; }
      acc[r4] = __builtin_amdgcn_wmma_f32_16x16x32_bf16(false, af, false, bf[kt],
                                                        (short)0, acc[r4], false, false);
    }
  }

  const int n  = wave * 16 + (lane & 15);
  const float bv = bias ? bias[n] : 0.0f;
#pragma unroll
  for (int r4 = 0; r4 < 4; ++r4) {
    const int m0 = (rt0 + r4) * 16 + 8 * (lane >> 4);
    float vals[8];
#pragma unroll
    for (int r = 0; r < 8; ++r) {
      float v = acc[r4][r] + bv;
      if (doRelu) v = fmaxf(v, 0.0f);
      vals[r] = v;
    }
    if (outF) {
#pragma unroll
      for (int r = 0; r < 8; ++r) outF[(size_t)(m0 + r) * D_ + n] = vals[r];
    }
    if (outB) {
#pragma unroll
      for (int r = 0; r < 8; ++r) outB[(size_t)(m0 + r) * D_ + n] = f2bf_bits(vals[r]);
    }
    if (outT) {  // transposed per-batch store: outT[b][n][m_local], 8 halves/lane
      int bidx = m0 / tokensPerBatch;
      int ml   = m0 - bidx * tokensPerBatch;
      Half8 pk;
#pragma unroll
      for (int r = 0; r < 8; ++r) pk.h[r] = f2bf(vals[r]);
      *(uint4*)((__bf16*)outT + ((size_t)(bidx * D_ + n)) * tokensPerBatch + ml) = pk.u;
    }
  }
}

// ------------------------------------------------------------------ attention
// 4 waves (128 threads) cooperate on one (b, h, 16-query tile).
// Wave w owns keys [w*512, (w+1)*512).
#define SROW (NK_ + 4)   // padded row stride (floats) to spread LDS banks

__global__ __launch_bounds__(128)
void attn_kernel(const uint16_t* __restrict__ qpu, const uint16_t* __restrict__ kpu,
                 const uint16_t* __restrict__ vTu, const unsigned char* __restrict__ mask,
                 float* __restrict__ o)
{
  __shared__ float S[16 * SROW];        // ~128.3 KB logits slab
  __shared__ float Opart[4][16][16];    // per-wave PV partials
  __shared__ float Red[16][8];          // softmax cross-segment reductions

  const __bf16* qp = (const __bf16*)qpu;
  const __bf16* kp = (const __bf16*)kpu;
  const __bf16* vT = (const __bf16*)vTu;

  const int tid  = threadIdx.x;
  const int lane = tid & 31;
  const int wave = tid >> 5;
  const int NQT = NQ_ / 16;
  int idx = blockIdx.x;
  const int qt = idx % NQT; idx /= NQT;
  const int h  = idx % H_;  idx /= H_;
  const int b  = idx;

  const int mrow = lane & 15;
  const int kb   = (lane >> 4) * 8;
  const float scale = 0.08838834764831845f;     // 1/sqrt(128)
  const float NEG_INF = -__builtin_inff();

  // Q A-fragment (16x32, K 16..31 zero-padded); identical in all 4 waves.
  v16bf qf;
  {
    Half8 a0, a1;
    const __bf16* qr = qp + ((size_t)(b * NQ_ + qt * 16 + mrow)) * D_ + h * DH_;
    a0.u = *(const uint4*)qr;
    a1.u = *(const uint4*)(qr + 8);
#pragma unroll
    for (int i = 0; i < 8; ++i) { qf[i] = lane < 16 ? a0.h[i] : (__bf16)0.0f; qf[8 + i] = lane < 16 ? a1.h[i] : (__bf16)0.0f; }
  }

  const unsigned char* mb = mask + (size_t)b * NK_;

  // ---- logits = (Q Kh^T) * scale, masked, into LDS (wave w: 32 k-tiles)
  for (int i = 0; i < 32; ++i) {
    const int k0 = wave * 512 + i * 16;
    v16bf kf;
    if (lane < 16) {   // B-frag rows K=0..15 valid; lanes 16..31 zero pad
      const __bf16* kr = kp + ((size_t)(b * NK_ + k0 + lane)) * D_ + h * DH_;
      Half8 b0, b1;
      b0.u = *(const uint4*)kr;
      b1.u = *(const uint4*)(kr + 8);
#pragma unroll
      for (int ii = 0; ii < 8; ++ii) { kf[ii] = b0.h[ii]; kf[8 + ii] = b1.h[ii]; }
      // prefetch next k-tile's K rows toward the WGP (L2-resident data)
      if (i < 31) __builtin_prefetch(kr + 16 * D_, 0, 1);
    } else {
#pragma unroll
      for (int ii = 0; ii < 16; ++ii) kf[ii] = (__bf16)0.0f;
    }
    v8f c = {};
    c = __builtin_amdgcn_wmma_f32_16x16x32_bf16(false, qf, false, kf,
                                                (short)0, c, false, false);
    const int kk = k0 + (lane & 15);
    const bool valid = mb[kk] != 0;
    const int mh = 8 * (lane >> 4);
#pragma unroll
    for (int r = 0; r < 8; ++r)
      S[(mh + r) * SROW + kk] = valid ? c[r] * scale : NEG_INF;
  }
  __syncthreads();

  // ---- softmax over k: 128 threads = 16 rows x 8 segments of 256 keys
  {
    const int row = tid & 15;
    const int seg = tid >> 4;                  // 0..7
    float* Srow = S + row * SROW;
    const int base = seg * 256;

    float mx = NEG_INF;
    for (int j = 0; j < 256; ++j) mx = fmaxf(mx, Srow[base + j]);
    Red[row][seg] = mx;
    __syncthreads();
    float m = NEG_INF;
#pragma unroll
    for (int s = 0; s < 8; ++s) m = fmaxf(m, Red[row][s]);
    if (!(m > NEG_INF)) m = 0.0f;              // fully-masked row -> probs 0
    __syncthreads();                            // Red reuse

    float sum = 0.0f;
    for (int j = 0; j < 256; ++j) {
      float e = __expf(Srow[base + j] - m);
      Srow[base + j] = e;
      sum += e;
    }
    Red[row][seg] = sum;
    __syncthreads();
    float tot = 0.0f;
#pragma unroll
    for (int s = 0; s < 8; ++s) tot += Red[row][s];
    float inv = tot > 0.0f ? 1.0f / tot : 0.0f;
    for (int j = 0; j < 256; ++j) Srow[base + j] *= inv;
  }
  __syncthreads();

  // ---- O = P @ Vh: wave w accumulates its 512-key span (16 WMMAs)
  v8f acc = {};
  for (int i = 0; i < 16; ++i) {
    const int k0 = wave * 512 + i * 32;
    v16bf pf;
    {
      const float* pr = S + (size_t)mrow * SROW + k0 + kb;
      float4 f0 = *(const float4*)(pr);
      float4 f1 = *(const float4*)(pr + 4);
      float4 f2 = *(const float4*)(pr + 16);
      float4 f3 = *(const float4*)(pr + 20);
      pf[0]  = f2bf(f0.x); pf[1]  = f2bf(f0.y); pf[2]  = f2bf(f0.z); pf[3]  = f2bf(f0.w);
      pf[4]  = f2bf(f1.x); pf[5]  = f2bf(f1.y); pf[6]  = f2bf(f1.z); pf[7]  = f2bf(f1.w);
      pf[8]  = f2bf(f2.x); pf[9]  = f2bf(f2.y); pf[10] = f2bf(f2.z); pf[11] = f2bf(f2.w);
      pf[12] = f2bf(f3.x); pf[13] = f2bf(f3.y); pf[14] = f2bf(f3.z); pf[15] = f2bf(f3.w);
    }
    v16bf vf;
    {
      const __bf16* vr = vT + ((size_t)(b * D_ + h * DH_ + (lane & 15))) * NK_
                            + k0 + (lane >> 4) * 8;
      Half8 b0, b1;
      b0.u = *(const uint4*)vr;
      b1.u = *(const uint4*)(vr + 16);
#pragma unroll
      for (int ii = 0; ii < 8; ++ii) { vf[ii] = b0.h[ii]; vf[8 + ii] = b1.h[ii]; }
      if (i < 15) __builtin_prefetch(vr + 32, 0, 1);   // next V chunk
    }
    acc = __builtin_amdgcn_wmma_f32_16x16x32_bf16(false, pf, false, vf,
                                                  (short)0, acc, false, false);
  }
  // stash per-wave partials, then cross-wave reduce
  {
    const int n  = lane & 15;
    const int m0 = 8 * (lane >> 4);
#pragma unroll
    for (int r = 0; r < 8; ++r) Opart[wave][m0 + r][n] = acc[r];
  }
  __syncthreads();
  for (int e = tid; e < 256; e += 128) {
    const int m = e >> 4, nn = e & 15;
    float v = Opart[0][m][nn] + Opart[1][m][nn] + Opart[2][m][nn] + Opart[3][m][nn];
    o[((size_t)(b * NQ_ + qt * 16 + m)) * D_ + h * DH_ + nn] = v;
  }
}

// -------------------------------------------------------------- residual + LN
__global__ __launch_bounds__(256)
void add_ln_kernel(const float* __restrict__ x, const float* __restrict__ y,
                   const float* __restrict__ g, const float* __restrict__ bb,
                   float* __restrict__ outF, uint16_t* __restrict__ outB)
{
  const int lane = threadIdx.x & 31;
  const int wave = threadIdx.x >> 5;
  const size_t row = (size_t)blockIdx.x * 8 + wave;
  const float4 xv = ((const float4*)(x + row * D_))[lane];
  const float4 yv = ((const float4*)(y + row * D_))[lane];
  float4 v = make_float4(xv.x + yv.x, xv.y + yv.y, xv.z + yv.z, xv.w + yv.w);
  float s  = v.x + v.y + v.z + v.w;
  float ss = v.x * v.x + v.y * v.y + v.z * v.z + v.w * v.w;
  for (int o = 16; o > 0; o >>= 1) {
    s  += __shfl_xor(s,  o, 32);
    ss += __shfl_xor(ss, o, 32);
  }
  const float mu  = s * (1.0f / D_);
  const float var = ss * (1.0f / D_) - mu * mu;
  const float rs  = rsqrtf(var + 1e-6f);
  const int c = lane * 4;
  float4 ov;
  ov.x = (v.x - mu) * rs * g[c + 0] + bb[c + 0];
  ov.y = (v.y - mu) * rs * g[c + 1] + bb[c + 1];
  ov.z = (v.z - mu) * rs * g[c + 2] + bb[c + 2];
  ov.w = (v.w - mu) * rs * g[c + 3] + bb[c + 3];
  ((float4*)(outF + row * D_))[lane] = ov;
  if (outB) {
    union { uint2 u; uint16_t s[4]; } pk;
    pk.s[0] = f2bf_bits(ov.x); pk.s[1] = f2bf_bits(ov.y);
    pk.s[2] = f2bf_bits(ov.z); pk.s[3] = f2bf_bits(ov.w);
    ((uint2*)(outB + row * D_))[lane] = pk.u;
  }
}

// ---------------------------------------------------------------------------
extern "C" void kernel_launch(void* const* d_in, const int* in_sizes, int n_in,
                              void* d_out, int out_size, void* d_ws, size_t ws_size,
                              hipStream_t stream) {
  (void)in_sizes; (void)n_in; (void)out_size; (void)ws_size;

  const float* q  = (const float*)d_in[0];
  const float* k  = (const float*)d_in[1];
  const float* v  = (const float*)d_in[2];
  const unsigned char* mask = (const unsigned char*)d_in[3];
  const float* Wq = (const float*)d_in[4];
  const float* bq = (const float*)d_in[5];
  const float* Wk = (const float*)d_in[6];
  const float* bk = (const float*)d_in[7];
  const float* Wv = (const float*)d_in[8];
  const float* bv = (const float*)d_in[9];
  const float* Wo = (const float*)d_in[10];
  const float* bo = (const float*)d_in[11];
  const float* g1 = (const float*)d_in[12];
  const float* b1 = (const float*)d_in[13];
  const float* g2 = (const float*)d_in[14];
  const float* b2 = (const float*)d_in[15];
  float* out = (float*)d_out;

  char* base = (char*)d_ws;
  size_t off = 0;
  auto take = [&](size_t bytes) -> char* {
    off = (off + 255) & ~(size_t)255;
    char* p = base + off;
    off += bytes;
    return p;
  };
  const size_t NTOK = (size_t)B_ * NQ_;          // 8192
  const size_t NELT = NTOK * D_;                 // 1,048,576

  uint16_t* q_bf   = (uint16_t*)take(NELT * 2);
  uint16_t* k_bf   = (uint16_t*)take(NELT * 2);
  uint16_t* v_bf   = (uint16_t*)take(NELT * 2);
  float*    qp_f   = (float*)   take(NELT * 4);
  uint16_t* qp_bf  = (uint16_t*)take(NELT * 2);
  uint16_t* kp_bf  = (uint16_t*)take(NELT * 2);
  uint16_t* vT_bf  = (uint16_t*)take(NELT * 2);
  uint16_t* Wq_pk  = (uint16_t*)take(16384 * 2);
  uint16_t* Wk_pk  = (uint16_t*)take(16384 * 2);
  uint16_t* Wv_pk  = (uint16_t*)take(16384 * 2);
  uint16_t* Wo_pk  = (uint16_t*)take(16384 * 2);
  float*    o_f    = (float*)   take(NELT * 4);
  float*    x1_f   = (float*)   take(NELT * 4);
  uint16_t* x1_bf  = (uint16_t*)take(NELT * 2);
  float*    y_f    = (float*)   take(NELT * 4);

  const int n = (int)NELT;
  cvt_f32_bf16<<<4096, 256, 0, stream>>>(q, q_bf, n);
  cvt_f32_bf16<<<4096, 256, 0, stream>>>(k, k_bf, n);
  cvt_f32_bf16<<<4096, 256, 0, stream>>>(v, v_bf, n);

  pack_w<<<64, 256, 0, stream>>>(Wq, Wq_pk);
  pack_w<<<64, 256, 0, stream>>>(Wk, Wk_pk);
  pack_w<<<64, 256, 0, stream>>>(Wv, Wv_pk);
  pack_w<<<64, 256, 0, stream>>>(Wo, Wo_pk);

  // qp (f32 + bf16), kp (bf16), vT (bf16 transposed per batch)
  gemm_bf16<<<128, 256, 0, stream>>>(q_bf, Wq_pk, bq, qp_f, qp_bf, nullptr, NK_, 0);
  gemm_bf16<<<128, 256, 0, stream>>>(k_bf, Wk_pk, bk, nullptr, kp_bf, nullptr, NK_, 0);
  gemm_bf16<<<128, 256, 0, stream>>>(v_bf, Wv_pk, bv, nullptr, nullptr, vT_bf, NK_, 0);

  attn_kernel<<<B_ * H_ * (NQ_ / 16), 128, 0, stream>>>(qp_bf, kp_bf, vT_bf, mask, o_f);

  add_ln_kernel<<<1024, 256, 0, stream>>>(qp_f, o_f, g1, b1, x1_f, x1_bf);

  gemm_bf16<<<128, 256, 0, stream>>>(x1_bf, Wo_pk, bo, y_f, nullptr, nullptr, NK_, 1);

  add_ln_kernel<<<1024, 256, 0, stream>>>(x1_f, y_f, g2, b2, out, nullptr);
}
